// MusicTransformer_56796647522580
// MI455X (gfx1250) — compile-verified, matrix operand order checked by reading, not run on hardware
//
#include <hip/hip_runtime.h>
#include <hip/hip_bf16.h>
#include <math.h>

// ---------------------------------------------------------------------------
// MusicTransformer forward for MI455X (gfx1250, wave32, WMMA + TDM).
// N=2, S=2048, E=512, H=8, HD=64, L=4, FE=2048.
// - all matmuls: v_wmma_f32_16x16x32_f16 (f16 in, fp32 accumulate)
// - all LDS tile staging: tensor_load_to_lds (TDM) with descriptor padding
//   so LDS rows land at conflict-friendly strides; double-buffered in GEMM
// - flash attention: scores never touch global memory
// ---------------------------------------------------------------------------

typedef __attribute__((ext_vector_type(16))) _Float16 v16h;
typedef __attribute__((ext_vector_type(8)))  _Float16 v8h;
typedef __attribute__((ext_vector_type(8)))  float    v8f;
typedef __attribute__((ext_vector_type(4)))  unsigned v4u;
typedef __attribute__((ext_vector_type(8)))  unsigned v8u;

#define WMMA_F16(A, B, C) \
  __builtin_amdgcn_wmma_f32_16x16x32_f16(false, (A), false, (B), (short)0, (C), false, false)

static constexpr int Ncfg  = 2;
static constexpr int Scfg  = 2048;
static constexpr int Ecfg  = 512;
static constexpr int Hcfg  = 8;
static constexpr int HDcfg = 64;
static constexpr int Lcfg  = 4;
static constexpr int FEcfg = 2048;

// ---------------------------------------------------------------------------
// TDM: load a 2-D f16 tile (rows x cols, row stride in elements) from global
// into LDS, with LDS padding after each row (pad fields per ISA 08 §8.4).
// Descriptor group0/group1 packing per CDNA5 ISA.
// Issued per-wave; caller guards so only wave 0 of the block issues.
// ---------------------------------------------------------------------------
__device__ __forceinline__ void tdm_load_tile_f16(unsigned ldsAddr,
                                                  const _Float16* g,
                                                  unsigned cols, unsigned rows,
                                                  unsigned rowStrideElems,
                                                  unsigned padInterval,
                                                  unsigned padAmount)
{
    unsigned long long ga = (unsigned long long)(size_t)g;
    v4u g0;
    g0.x = 1u;                                              // count=1 (valid D#)
    g0.y = ldsAddr;                                         // LDS byte address
    g0.z = (unsigned)ga;                                    // global_addr[31:0]
    g0.w = (unsigned)((ga >> 32) & 0x01FFFFFFull) | (2u << 30); // addr[56:32] | type=2
    v8u g1;
    g1[0] = (1u << 16)                 // data_size = 1 -> 2 bytes (f16)
          | (1u << 20)                 // pad_enable
          | (padInterval << 22)        // pad_interval (2^(v+1) DWORDs)
          | (padAmount << 25);         // pad_amount ((v+1) DWORDs)
    g1[1] = cols << 16;                // tensor_dim0[15:0]
    g1[2] = rows << 16;                // tensor_dim0[31:16]=0 | tensor_dim1[15:0]
    g1[3] = cols << 16;                // tensor_dim1[31:16]=0 | tile_dim0
    g1[4] = rows;                      // tile_dim1 | tile_dim2=0
    g1[5] = rowStrideElems;            // tensor_dim0_stride[31:0]
    g1[6] = 0u;                        // stride0[47:32] | stride1[15:0]
    g1[7] = 0u;                        // stride1[47:16]
    // 2-D tensor: groups 2/3 are NULL -> two-operand form.
    asm volatile("tensor_load_to_lds %0, %1" :: "s"(g0), "s"(g1) : "memory");
}

// ---------------------------------------------------------------------------
// Embedding: out[n,s,e] = emb_w[tok[n,s], e] + pos_w[s, e]  (fp32 + f16 copies)
// ---------------------------------------------------------------------------
__global__ void embed_kernel(const long long* __restrict__ tok,
                             const float* __restrict__ emb,
                             const float* __restrict__ pos,
                             float* __restrict__ outF,
                             _Float16* __restrict__ outH)
{
    size_t idx = (size_t)blockIdx.x * 256 + threadIdx.x;    // over N*S*E
    int e      = (int)(idx & (Ecfg - 1));
    size_t ns  = idx >> 9;
    int s      = (int)(ns & (Scfg - 1));
    long long t = tok[ns];
    float v = emb[(size_t)t * Ecfg + e] + pos[(size_t)s * Ecfg + e];
    outF[idx] = v;
    outH[idx] = (_Float16)v;
}

// ---------------------------------------------------------------------------
// Weight convert+transpose: dst[N][K] (f16) = src[K][N] (fp32). Once/launch.
// ---------------------------------------------------------------------------
__global__ void wtrans_kernel(const float* __restrict__ src,
                              _Float16* __restrict__ dst, int K, int Nn)
{
    size_t idx = (size_t)blockIdx.x * 256 + threadIdx.x;
    if (idx >= (size_t)K * Nn) return;
    int k = (int)(idx / Nn);
    int n = (int)(idx - (size_t)k * Nn);
    dst[(size_t)n * K + k] = (_Float16)src[idx];
}

// ---------------------------------------------------------------------------
// V transpose per layer: vT[(n*H+h)*64 + d][s] = v[((n*S+s)*H+h)*64 + d]
// so attention's V tiles are row-contiguous (TDM-loadable).
// ---------------------------------------------------------------------------
__global__ void vtrans_kernel(const _Float16* __restrict__ v,
                              _Float16* __restrict__ vT)
{
    size_t idx = (size_t)blockIdx.x * 256 + threadIdx.x;    // over N*H*HD*S
    int s   = (int)(idx & (Scfg - 1));
    size_t r = idx >> 11;                                   // (n*H+h)*64 + d
    int d   = (int)(r & 63);
    int nh  = (int)(r >> 6);
    int n   = nh >> 3, h = nh & 7;
    vT[idx] = v[((size_t)(n * Scfg + s) * Hcfg + h) * HDcfg + d];
}

// ---------------------------------------------------------------------------
// WMMA GEMM:  C[M,Nn] = A[M,K] * BT[Nn,K]^T  (+bias) (+relu)
// A, BT are f16 row-major in global. Block tile 128x64, BK=32, 8 waves,
// each wave computes 32x32 (4 WMMAs/K-step). Both tiles staged by TDM with
// LDS padding to stride GTS=48 halfs; double-buffered across K-steps.
// Requires M%128==0, Nn%64==0, K%32==0.
// ---------------------------------------------------------------------------
#define GTS 48   // LDS row stride in halfs (96B; keeps b128 fragment reads aligned)

__global__ void gemm_wmma_f16(const _Float16* __restrict__ A,
                              const _Float16* __restrict__ BT,
                              const float* __restrict__ bias,
                              float* __restrict__ Cf,
                              _Float16* __restrict__ Ch,
                              int M, int Nn, int K, int relu)
{
    __shared__ __align__(16) _Float16 As[2][128 * GTS];
    __shared__ __align__(16) _Float16 Bs[2][64 * GTS];

    const int t      = threadIdx.x;
    const int lane   = t & 31;
    const int wave   = t >> 5;
    const int wm     = wave & 3;          // M-group (32 rows)
    const int wn     = wave >> 2;         // N-group (32 cols)
    const int bm     = blockIdx.y * 128;
    const int bn     = blockIdx.x * 64;
    const int half16 = lane >> 4;
    const int kb     = half16 * 8;
    const int mr     = lane & 15;

    const _Float16* Abase = A  + (size_t)bm * K;
    const _Float16* Bbase = BT + (size_t)bn * K;
    const int nk = K >> 5;

    // rows of 32 halfs = 16 DWORDs -> pad_interval=3 (16 DW); pad 8 DW = 16 halfs
    if (t < 32) {
        tdm_load_tile_f16((unsigned)(size_t)&As[0][0], Abase, 32, 128, (unsigned)K, 3, 7);
        tdm_load_tile_f16((unsigned)(size_t)&Bs[0][0], Bbase, 32, 64,  (unsigned)K, 3, 7);
    }

    v8f acc[2][2] = {{{}, {}}, {{}, {}}};

    for (int ki = 0; ki < nk; ++ki) {
        if (t < 32) {
            if (ki + 1 < nk) {
                int nb = (ki + 1) & 1;
                const _Float16* an = Abase + ((size_t)(ki + 1) << 5);
                const _Float16* bn2 = Bbase + ((size_t)(ki + 1) << 5);
                tdm_load_tile_f16((unsigned)(size_t)&As[nb][0], an,  32, 128, (unsigned)K, 3, 7);
                tdm_load_tile_f16((unsigned)(size_t)&Bs[nb][0], bn2, 32, 64,  (unsigned)K, 3, 7);
                __builtin_amdgcn_s_wait_tensorcnt(2);   // current stage's 2 DMAs done
            } else {
                __builtin_amdgcn_s_wait_tensorcnt(0);
            }
        }
        __syncthreads();

        const _Float16* as = &As[ki & 1][0];
        const _Float16* bs = &Bs[ki & 1][0];

        v16h a0, a1, b0, b1;
        {
            int row = wm * 32 + mr;
            v8h lo = *(const v8h*)&as[row * GTS + kb];
            v8h hi = *(const v8h*)&as[row * GTS + 16 + kb];
#pragma unroll
            for (int j = 0; j < 8; ++j) { a0[j] = lo[j]; a0[8 + j] = hi[j]; }
            row += 16;
            lo = *(const v8h*)&as[row * GTS + kb];
            hi = *(const v8h*)&as[row * GTS + 16 + kb];
#pragma unroll
            for (int j = 0; j < 8; ++j) { a1[j] = lo[j]; a1[8 + j] = hi[j]; }
        }
        {
            int col = wn * 32 + mr;
            v8h lo = *(const v8h*)&bs[col * GTS + kb];
            v8h hi = *(const v8h*)&bs[col * GTS + 16 + kb];
#pragma unroll
            for (int j = 0; j < 8; ++j) { b0[j] = lo[j]; b0[8 + j] = hi[j]; }
            col += 16;
            lo = *(const v8h*)&bs[col * GTS + kb];
            hi = *(const v8h*)&bs[col * GTS + 16 + kb];
#pragma unroll
            for (int j = 0; j < 8; ++j) { b1[j] = lo[j]; b1[8 + j] = hi[j]; }
        }

        acc[0][0] = WMMA_F16(a0, b0, acc[0][0]);
        acc[0][1] = WMMA_F16(a0, b1, acc[0][1]);
        acc[1][0] = WMMA_F16(a1, b0, acc[1][0]);
        acc[1][1] = WMMA_F16(a1, b1, acc[1][1]);
        __syncthreads();
    }

    // epilogue: C-layout (VGPR i: M = i + 8*half16, N = mr)
#pragma unroll
    for (int mi = 0; mi < 2; ++mi)
#pragma unroll
        for (int ni = 0; ni < 2; ++ni) {
            int gn = bn + wn * 32 + ni * 16 + mr;
            float bb = bias ? bias[gn] : 0.0f;
#pragma unroll
            for (int i = 0; i < 8; ++i) {
                int gm = bm + wm * 32 + mi * 16 + i + 8 * half16;
                float val = acc[mi][ni][i] + bb;
                if (relu) val = fmaxf(val, 0.0f);
                if (Cf) Cf[(size_t)gm * Nn + gn] = val;
                if (Ch) Ch[(size_t)gm * Nn + gn] = (_Float16)val;
            }
        }
}

// ---------------------------------------------------------------------------
// Flash attention. Grid (S/128, N*H), block 256 (8 waves), wave = 16-row strip.
// Q[N,S,H,HD], K[N,S,H,HD] f16; VT[(n*H+h)*64+d][S] f16. Output f16 [N,S,E].
// Q/K/V tiles staged via TDM (LDS stride 72 halfs: rows of 64 halfs = 32 DW ->
// pad_interval=4, pad 4 DW = 8 halfs).
// ---------------------------------------------------------------------------
#define AQS 72

__global__ void attn_flash_wmma(const _Float16* __restrict__ Q,
                                const _Float16* __restrict__ Kg,
                                const _Float16* __restrict__ VT,
                                const int*      __restrict__ mask,
                                _Float16* __restrict__ O,
                                float invScale)
{
    __shared__ __align__(16) _Float16 Qs[128 * AQS];
    __shared__ __align__(16) _Float16 Ks[64 * AQS];
    __shared__ __align__(16) _Float16 Vs[64 * AQS];        // Vs[d][key]
    __shared__ __align__(16) _Float16 Ps[8 * 16 * AQS];

    const int t      = threadIdx.x;
    const int lane   = t & 31;
    const int wave   = t >> 5;
    const int nh     = blockIdx.y;
    const int n      = nh >> 3;
    const int h      = nh & 7;
    const int q0     = blockIdx.x * 128;
    const int half16 = lane >> 4;
    const int kb     = half16 * 8;
    const int mr     = lane & 15;

    const _Float16* qbase = Q  + ((size_t)(n * Scfg + q0) * Hcfg + h) * HDcfg;
    const _Float16* kbase = Kg + ((size_t)(n * Scfg) * Hcfg + h) * HDcfg;
    const _Float16* vbase = VT + (size_t)nh * HDcfg * Scfg;

    if (t < 32)
        tdm_load_tile_f16((unsigned)(size_t)&Qs[0], qbase, 64, 128, Ecfg, 4, 3);

    v8f acc[4] = {{}, {}, {}, {}};
    float mrow[8], lrow[8];
#pragma unroll
    for (int i = 0; i < 8; ++i) { mrow[i] = -3.0e38f; lrow[i] = 0.0f; }

    for (int kblk = 0; kblk < Scfg; kblk += 64) {
        if (t < 32) {
            tdm_load_tile_f16((unsigned)(size_t)&Ks[0],
                              kbase + (size_t)kblk * Ecfg, 64, 64, Ecfg, 4, 3);
            tdm_load_tile_f16((unsigned)(size_t)&Vs[0],
                              vbase + kblk, 64, 64, Scfg, 4, 3);
            __builtin_amdgcn_s_wait_tensorcnt(0);
        }
        __syncthreads();

        // ---- scores: Q(16x64) * K^T(64x64) -> 4 tiles, 2 K-steps ----
        v8f sc[4] = {{}, {}, {}, {}};
#pragma unroll
        for (int ks = 0; ks < 64; ks += 32) {
            v16h afrag;
            {
                int row = wave * 16 + mr;
                v8h lo = *(const v8h*)&Qs[row * AQS + ks + kb];
                v8h hi = *(const v8h*)&Qs[row * AQS + ks + 16 + kb];
#pragma unroll
                for (int j = 0; j < 8; ++j) { afrag[j] = lo[j]; afrag[8 + j] = hi[j]; }
            }
#pragma unroll
            for (int jt = 0; jt < 4; ++jt) {
                v16h bfrag;
                int krow = jt * 16 + mr;
                v8h lo = *(const v8h*)&Ks[krow * AQS + ks + kb];
                v8h hi = *(const v8h*)&Ks[krow * AQS + ks + 16 + kb];
#pragma unroll
                for (int j = 0; j < 8; ++j) { bfrag[j] = lo[j]; bfrag[8 + j] = hi[j]; }
                sc[jt] = WMMA_F16(afrag, bfrag, sc[jt]);
            }
        }

        // ---- streaming softmax ----
        float p[4][8];
#pragma unroll
        for (int i = 0; i < 8; ++i) {
            float e[4];
            float rmax = -3.0e38f;
#pragma unroll
            for (int jt = 0; jt < 4; ++jt) {
                int key = kblk + jt * 16 + mr;
                float ev = sc[jt][i] * invScale;
                if (mask[n * Scfg + key] == 0) ev = -1.0e20f;
                e[jt] = ev;
                rmax = fmaxf(rmax, ev);
            }
#pragma unroll
            for (int off = 1; off < 16; off <<= 1)
                rmax = fmaxf(rmax, __shfl_xor(rmax, off, 32));
            float mnew = fmaxf(mrow[i], rmax);
            float corr = __expf(mrow[i] - mnew);
            mrow[i] = mnew;
            float rs = 0.0f;
#pragma unroll
            for (int jt = 0; jt < 4; ++jt) {
                float pv = __expf(e[jt] - mnew);
                p[jt][i] = pv;
                rs += pv;
            }
#pragma unroll
            for (int off = 1; off < 16; off <<= 1)
                rs += __shfl_xor(rs, off, 32);
            lrow[i] = lrow[i] * corr + rs;
#pragma unroll
            for (int jt = 0; jt < 4; ++jt) acc[jt][i] *= corr;
        }

        // ---- P (C-layout) -> per-wave LDS scratch as f16 ----
#pragma unroll
        for (int jt = 0; jt < 4; ++jt)
#pragma unroll
            for (int i = 0; i < 8; ++i) {
                int m = i + 8 * half16;
                Ps[(wave * 16 + m) * AQS + jt * 16 + mr] = (_Float16)p[jt][i];
            }
        __syncthreads();

        // ---- O += P(16x64) * V(64x64) ----
#pragma unroll
        for (int ks = 0; ks < 64; ks += 32) {
            v16h afrag;
            {
                int row = wave * 16 + mr;
                v8h lo = *(const v8h*)&Ps[row * AQS + ks + kb];
                v8h hi = *(const v8h*)&Ps[row * AQS + ks + 16 + kb];
#pragma unroll
                for (int j = 0; j < 8; ++j) { afrag[j] = lo[j]; afrag[8 + j] = hi[j]; }
            }
#pragma unroll
            for (int jd = 0; jd < 4; ++jd) {
                v16h bfrag;
                int drow = jd * 16 + mr;
                v8h lo = *(const v8h*)&Vs[drow * AQS + ks + kb];
                v8h hi = *(const v8h*)&Vs[drow * AQS + ks + 16 + kb];
#pragma unroll
                for (int j = 0; j < 8; ++j) { bfrag[j] = lo[j]; bfrag[8 + j] = hi[j]; }
                acc[jd] = WMMA_F16(afrag, bfrag, acc[jd]);
            }
        }
        __syncthreads();
    }

    // ---- normalize and store (f16) ----
#pragma unroll
    for (int jd = 0; jd < 4; ++jd)
#pragma unroll
        for (int i = 0; i < 8; ++i) {
            int m  = i + 8 * half16;
            int qr = q0 + wave * 16 + m;
            float val = acc[jd][i] / lrow[i];
            O[((size_t)(n * Scfg + qr) * Hcfg + h) * HDcfg + jd * 16 + mr] = (_Float16)val;
        }
}

// ---------------------------------------------------------------------------
// out = LayerNorm(A + B) * s + b   (row length 512); fp32 + f16 outputs.
// ---------------------------------------------------------------------------
__global__ void add_ln_kernel(const float* __restrict__ A,
                              const float* __restrict__ B,
                              const float* __restrict__ sc,
                              const float* __restrict__ bi,
                              float* __restrict__ OutF,
                              _Float16* __restrict__ OutH)
{
    __shared__ float red[256];
    const int row = blockIdx.x;
    const int t   = threadIdx.x;
    const float* a = A + (size_t)row * Ecfg;
    const float* b = B + (size_t)row * Ecfg;

    float x0 = a[t] + b[t];
    float x1 = a[t + 256] + b[t + 256];

    red[t] = x0 + x1;
    __syncthreads();
    for (int o = 128; o > 0; o >>= 1) {
        if (t < o) red[t] += red[t + o];
        __syncthreads();
    }
    float mean = red[0] * (1.0f / Ecfg);
    __syncthreads();

    float d0 = x0 - mean, d1 = x1 - mean;
    red[t] = d0 * d0 + d1 * d1;
    __syncthreads();
    for (int o = 128; o > 0; o >>= 1) {
        if (t < o) red[t] += red[t + o];
        __syncthreads();
    }
    float rstd = rsqrtf(red[0] * (1.0f / Ecfg) + 1e-5f);

    float v0 = d0 * rstd * sc[t]       + bi[t];
    float v1 = d1 * rstd * sc[t + 256] + bi[t + 256];
    OutF[(size_t)row * Ecfg + t]       = v0;
    OutF[(size_t)row * Ecfg + t + 256] = v1;
    OutH[(size_t)row * Ecfg + t]       = (_Float16)v0;
    OutH[(size_t)row * Ecfg + t + 256] = (_Float16)v1;
}

// ---------------------------------------------------------------------------
// pooled[n,e] = mean_s out[n,s,e];  out[n,j] = pooled @ fc_w + fc_b
// ---------------------------------------------------------------------------
__global__ void pool_kernel(const float* __restrict__ X, float* __restrict__ P)
{
    int idx = blockIdx.x * 256 + threadIdx.x;
    int n = idx >> 9, e = idx & (Ecfg - 1);
    float sum = 0.0f;
    for (int s = 0; s < Scfg; ++s)
        sum += X[((size_t)n * Scfg + s) * Ecfg + e];
    P[idx] = sum * (1.0f / Scfg);
}

__global__ void fc_kernel(const float* __restrict__ P,
                          const float* __restrict__ W,
                          const float* __restrict__ B,
                          float* __restrict__ Out)
{
    int idx = threadIdx.x;
    int n = idx >> 7, j = idx & 127;
    float sum = B[j];
    for (int e = 0; e < Ecfg; ++e)
        sum += P[n * Ecfg + e] * W[e * 128 + j];
    Out[idx] = sum;
}

// ---------------------------------------------------------------------------
// Host-side launch
// ---------------------------------------------------------------------------
extern "C" void kernel_launch(void* const* d_in, const int* in_sizes, int n_in,
                              void* d_out, int out_size, void* d_ws, size_t ws_size,
                              hipStream_t stream)
{
    (void)in_sizes; (void)n_in; (void)out_size; (void)ws_size;

    const long long* x     = (const long long*)d_in[0];
    const int*       mask  = (const int*)      d_in[1];
    const float*     emb_w = (const float*)    d_in[2];
    const float*     pos_w = (const float*)    d_in[3];
    const float*     wq    = (const float*)    d_in[4];
    const float*     wk    = (const float*)    d_in[5];
    const float*     wv    = (const float*)    d_in[6];
    const float*     wo    = (const float*)    d_in[7];
    const float*     bo    = (const float*)    d_in[8];
    const float*     ln1_s = (const float*)    d_in[9];
    const float*     ln1_b = (const float*)    d_in[10];
    const float*     ln2_s = (const float*)    d_in[11];
    const float*     ln2_b = (const float*)    d_in[12];
    const float*     w1    = (const float*)    d_in[13];
    const float*     b1    = (const float*)    d_in[14];
    const float*     w2    = (const float*)    d_in[15];
    const float*     b2    = (const float*)    d_in[16];
    const float*     fc_w  = (const float*)    d_in[17];
    const float*     fc_b  = (const float*)    d_in[18];

    const size_t SE = (size_t)Ncfg * Scfg * Ecfg;        // 2,097,152 elements

    char* p = (char*)d_ws;
    auto allocF = [&](size_t n) { float* r = (float*)p;    p += n * sizeof(float);    return r; };
    auto allocH = [&](size_t n) { _Float16* r = (_Float16*)p; p += n * sizeof(_Float16); return r; };

    float*    outF  = allocF(SE);
    float*    woF   = allocF(SE);
    float*    x1F   = allocF(SE);
    float*    ff2F  = allocF(SE);
    float*    pooled= allocF(Ncfg * Ecfg);
    _Float16* outH  = allocH(SE);
    _Float16* qH    = allocH(SE);
    _Float16* kH    = allocH(SE);
    _Float16* vH    = allocH(SE);
    _Float16* vT    = allocH(SE);
    _Float16* attH  = allocH(SE);
    _Float16* x1H   = allocH(SE);
    _Float16* ff1H  = allocH((size_t)Ncfg * Scfg * FEcfg);
    _Float16* wqT   = allocH((size_t)Lcfg * HDcfg * HDcfg);
    _Float16* wkT   = allocH((size_t)Lcfg * HDcfg * HDcfg);
    _Float16* wvT   = allocH((size_t)Lcfg * HDcfg * HDcfg);
    _Float16* woT   = allocH((size_t)Lcfg * Ecfg * Ecfg);
    _Float16* w1T   = allocH((size_t)Lcfg * Ecfg * FEcfg);
    _Float16* w2T   = allocH((size_t)Lcfg * FEcfg * Ecfg);

    const int M_tok = Ncfg * Scfg;                       // 4096
    const int M_qkv = Ncfg * Scfg * Hcfg;                // 32768
    const float invScale = 1.0f / sqrtf((float)Ecfg);

    // ---- one-time (per launch) weight convert+transpose to f16 [N][K] ----
    for (int i = 0; i < Lcfg; ++i) {
        size_t oS = (size_t)i * HDcfg * HDcfg;
        wtrans_kernel<<<(HDcfg * HDcfg + 255) / 256, 256, 0, stream>>>(wq + oS, wqT + oS, HDcfg, HDcfg);
        wtrans_kernel<<<(HDcfg * HDcfg + 255) / 256, 256, 0, stream>>>(wk + oS, wkT + oS, HDcfg, HDcfg);
        wtrans_kernel<<<(HDcfg * HDcfg + 255) / 256, 256, 0, stream>>>(wv + oS, wvT + oS, HDcfg, HDcfg);
        size_t oE = (size_t)i * Ecfg * Ecfg;
        wtrans_kernel<<<(Ecfg * Ecfg) / 256, 256, 0, stream>>>(wo + oE, woT + oE, Ecfg, Ecfg);
        size_t oW = (size_t)i * Ecfg * FEcfg;
        wtrans_kernel<<<(Ecfg * FEcfg) / 256, 256, 0, stream>>>(w1 + oW, w1T + oW, Ecfg, FEcfg);
        wtrans_kernel<<<(Ecfg * FEcfg) / 256, 256, 0, stream>>>(w2 + oW, w2T + oW, FEcfg, Ecfg);
    }

    // ---- embedding + positional ----
    embed_kernel<<<(unsigned)(SE / 256), 256, 0, stream>>>(x, emb_w, pos_w, outF, outH);

    // ---- transformer layers ----
    for (int i = 0; i < Lcfg; ++i) {
        const _Float16* wqT_i = wqT + (size_t)i * HDcfg * HDcfg;
        const _Float16* wkT_i = wkT + (size_t)i * HDcfg * HDcfg;
        const _Float16* wvT_i = wvT + (size_t)i * HDcfg * HDcfg;
        const _Float16* woT_i = woT + (size_t)i * Ecfg * Ecfg;
        const _Float16* w1T_i = w1T + (size_t)i * Ecfg * FEcfg;
        const _Float16* w2T_i = w2T + (size_t)i * FEcfg * Ecfg;
        const float* bo_i = bo + (size_t)i * Ecfg;
        const float* b1_i = b1 + (size_t)i * FEcfg;
        const float* b2_i = b2 + (size_t)i * Ecfg;
        const float* l1s  = ln1_s + (size_t)i * Ecfg;
        const float* l1b  = ln1_b + (size_t)i * Ecfg;
        const float* l2s  = ln2_s + (size_t)i * Ecfg;
        const float* l2b  = ln2_b + (size_t)i * Ecfg;

        // Q/K/V projections: [32768,64] @ [64,64]
        gemm_wmma_f16<<<dim3(1, M_qkv / 128), 256, 0, stream>>>(
            outH, wqT_i, nullptr, nullptr, qH, M_qkv, HDcfg, HDcfg, 0);
        gemm_wmma_f16<<<dim3(1, M_qkv / 128), 256, 0, stream>>>(
            outH, wkT_i, nullptr, nullptr, kH, M_qkv, HDcfg, HDcfg, 0);
        gemm_wmma_f16<<<dim3(1, M_qkv / 128), 256, 0, stream>>>(
            outH, wvT_i, nullptr, nullptr, vH, M_qkv, HDcfg, HDcfg, 0);

        // V -> [nh][d][s] for TDM-friendly attention tiles
        vtrans_kernel<<<(unsigned)(SE / 256), 256, 0, stream>>>(vH, vT);

        // flash attention
        attn_flash_wmma<<<dim3(Scfg / 128, Ncfg * Hcfg), 256, 0, stream>>>(
            qH, kH, vT, mask, attH, invScale);

        // output projection + bias
        gemm_wmma_f16<<<dim3(Ecfg / 64, M_tok / 128), 256, 0, stream>>>(
            attH, woT_i, bo_i, woF, nullptr, M_tok, Ecfg, Ecfg, 0);

        // x1 = LN(att_out + residual)
        add_ln_kernel<<<M_tok, 256, 0, stream>>>(woF, outF, l1s, l1b, x1F, x1H);

        // FFN1 (+ReLU), f16 out
        gemm_wmma_f16<<<dim3(FEcfg / 64, M_tok / 128), 256, 0, stream>>>(
            x1H, w1T_i, b1_i, nullptr, ff1H, M_tok, FEcfg, Ecfg, 1);

        // FFN2, fp32 out
        gemm_wmma_f16<<<dim3(Ecfg / 64, M_tok / 128), 256, 0, stream>>>(
            ff1H, w2T_i, b2_i, ff2F, nullptr, M_tok, Ecfg, FEcfg, 0);

        // out = LN(ff + x1)
        add_ln_kernel<<<M_tok, 256, 0, stream>>>(ff2F, x1F, l2s, l2b, outF, outH);
    }

    // ---- mean-pool + final FC -> [N,128] ----
    pool_kernel<<<(Ncfg * Ecfg) / 256, 256, 0, stream>>>(outF, pooled);
    fc_kernel<<<1, 256, 0, stream>>>(pooled, fc_w, fc_b, (float*)d_out);
}
